// HypergraphImputer_47450798686453
// MI455X (gfx1250) — matrix operand order repaired and checked by reading, not compiled, not run on hardware
//
#include <hip/hip_runtime.h>
#include <hip/hip_bf16.h>
#include <math.h>

// Problem constants (match reference)
#define BB 1024
#define KK 12
#define RR 8
#define DD 512
#define HH 2
#define EE 64
#define CTX 64
#define DH (DD / HH)   // 256

// ---------- WMMA types ----------
typedef __bf16 v16bf __attribute__((ext_vector_type(16)));
typedef float  v8f   __attribute__((ext_vector_type(8)));

union BFrag { v16bf v; unsigned short u[16]; };

__device__ __forceinline__ unsigned short f2bf(float x) {
    unsigned int u = __float_as_uint(x);
    u += 0x7FFFu + ((u >> 16) & 1u);   // round-to-nearest-even
    return (unsigned short)(u >> 16);
}

__device__ __forceinline__ float gelu_exact(float x) {
    return 0.5f * x * (1.0f + erff(x * 0.70710678118654752f));
}

// ---------- CDNA5 async global->LDS copy (ASYNCcnt-tracked, 16B/lane) ----------
// Generic pointers to __shared__ carry the LDS byte offset in bits[31:0]
// (ISA 10.2); that offset is the VDST operand.
__device__ __forceinline__ void async_copy_b128(void* lds_ptr, const void* gptr) {
    unsigned lds_off = (unsigned)(unsigned long long)lds_ptr;
    asm volatile("global_load_async_to_lds_b128 %0, %1, off"
                 :: "v"(lds_off), "v"((unsigned long long)gptr)
                 : "memory");
}
__device__ __forceinline__ void wait_async0() {
    asm volatile("s_wait_asynccnt 0x0" ::: "memory");
}

// Load A fragment (16x32 bf16, row-major source, contiguous k order, same as B):
// row = lane%16 ; K = 16*(lane/16) + j
__device__ __forceinline__ BFrag load_a_frag(const unsigned short* __restrict__ A,
                                             int lda, int row0, int k0, int lane, int Mclamp) {
    BFrag f;
    int m = row0 + (lane & 15);
    if (m >= Mclamp) m = Mclamp - 1;
    const unsigned short* p = A + (long)m * lda + k0 + ((lane >> 4) << 4);
#pragma unroll
    for (int j = 0; j < 16; ++j) f.u[j] = p[j];
    return f;
}

// Load B fragment for A@W^T: B(k,n) = W[n0+n][k0+k], W row-major [N, ldw].
// col n = lane%16 ; K = 16*(lane/16) + j
__device__ __forceinline__ BFrag load_b_frag(const unsigned short* __restrict__ W,
                                             int ldw, int n0, int k0, int lane) {
    BFrag f;
    const unsigned short* p = W + (long)(n0 + (lane & 15)) * ldw + k0 + ((lane >> 4) << 4);
#pragma unroll
    for (int j = 0; j < 16; ++j) f.u[j] = p[j];
    return f;
}

// ---------------------------------------------------------------------------
// k_prep: per-batch leave-one-out counts, has_src, pooled, X -> bf16
// grid = B, block = 256
// ---------------------------------------------------------------------------
__global__ void k_prep(const float* __restrict__ X, const float* __restrict__ Mctx,
                       float* __restrict__ counts, float* __restrict__ has_src,
                       float* __restrict__ pooled, unsigned short* __restrict__ Xbf) {
    __shared__ float sM[KK];
    __shared__ float sCnt[KK];
    int b = blockIdx.x, tid = threadIdx.x;
    if (tid < KK) sM[tid] = Mctx[b * KK + tid];
    __syncthreads();
    if (tid < KK) {
        float c = 0.f;
#pragma unroll
        for (int t = 0; t < KK; ++t) if (t != tid) c += sM[t];
        has_src[b * KK + tid] = (c > 0.f) ? 1.f : 0.f;
        float cc = (c < 1.f) ? 1.f : c;
        sCnt[tid] = cc;
        counts[b * KK + tid] = cc;
    }
    __syncthreads();
#pragma unroll
    for (int dd = 0; dd < 2; ++dd) {
        int d = tid + dd * 256;
        float x[KK];
#pragma unroll
        for (int t = 0; t < KK; ++t) {
            x[t] = X[((long)(b * KK + t)) * DD + d];
            Xbf[((long)(b * KK + t)) * DD + d] = f2bf(x[t]);
        }
#pragma unroll
        for (int s = 0; s < KK; ++s) {
            float a = 0.f;
#pragma unroll
            for (int t = 0; t < KK; ++t) if (t != s) a += sM[t] * x[t];
            pooled[((long)(b * KK + s)) * DD + d] = a / sCnt[s];
        }
    }
}

// ---------------------------------------------------------------------------
// k_conv2d: strided fp32 -> bf16 (rows x cols, src row stride sstride)
// ---------------------------------------------------------------------------
__global__ void k_conv2d(const float* __restrict__ src, unsigned short* __restrict__ dst,
                         int rows, int cols, int sstride) {
    long i = (long)blockIdx.x * 256 + threadIdx.x;
    long n = (long)rows * cols;
    if (i < n) {
        int rr = (int)(i / cols), cc = (int)(i % cols);
        dst[i] = f2bf(src[(long)rr * sstride + cc]);
    }
}

// ---------------------------------------------------------------------------
// k_tpb: tpb[r][d] = b1[d] + sum_e edge_tokens[r][e] * W1[d][DD+e]
// grid = R, block = 256
// ---------------------------------------------------------------------------
__global__ void k_tpb(const float* __restrict__ edge_tokens, const float* __restrict__ W1,
                      const float* __restrict__ b1, float* __restrict__ tpb) {
    __shared__ float sE[EE];
    int r = blockIdx.x, tid = threadIdx.x;
    if (tid < EE) sE[tid] = edge_tokens[r * EE + tid];
    __syncthreads();
#pragma unroll
    for (int dd = 0; dd < 2; ++dd) {
        int d = tid + dd * 256;
        float a = b1[d];
        const float* w = W1 + (long)d * (DD + EE) + DD;
#pragma unroll 8
        for (int e = 0; e < EE; ++e) a += sE[e] * w[e];
        tpb[r * DD + d] = a;
    }
}

// ---------------------------------------------------------------------------
// k_gemm_bf16: C[M,N] = alpha * (A[M,Kd] @ W[N,Kd]^T) * rowscale[row]?
// A tile (16 x Kd) staged once per block in LDS via async-to-LDS;
// B fragments double-buffered from global. Kd must be <= 512 (here: 512).
// block = 256 (8 waves = 128 cols); grid = ((N+127)/128, (M+15)/16)
// ---------------------------------------------------------------------------
__global__ void k_gemm_bf16(const unsigned short* __restrict__ A, int lda,
                            const unsigned short* __restrict__ W, int ldw,
                            float* __restrict__ Cmat, int ldc,
                            int M, int N, int Kd, float alpha,
                            const float* __restrict__ rowscale) {
    __shared__ __align__(16) unsigned short sA[16 * 512];
    int tid = threadIdx.x;
    int lane = tid & 31;
    int wave = tid >> 5;
    int row0 = blockIdx.y * 16;

    // ---- async stage of the block-shared A tile (16 rows x Kd) ----
    int chunks = (16 * Kd) >> 3;              // 16-byte chunks
    for (int i = tid; i < chunks; i += 256) {
        int idx = i << 3;
        int m = idx / Kd, c = idx % Kd;
        int gm = row0 + m; if (gm >= M) gm = M - 1;
        async_copy_b128(&sA[m * Kd + c], A + (long)gm * lda + c);
    }
    wait_async0();
    __syncthreads();

    int n0 = (blockIdx.x * 8 + wave) * 16;
    if (n0 >= N || row0 >= M) return;

    v8f acc = {};
    BFrag bcur = load_b_frag(W, ldw, n0, 0, lane);
    for (int k0 = 0; k0 < Kd; k0 += 32) {
        int kn = (k0 + 32 < Kd) ? (k0 + 32) : 0;       // double buffer (last is dummy)
        BFrag bnext = load_b_frag(W, ldw, n0, kn, lane);
        if (k0 + 64 < Kd)
            __builtin_prefetch(W + (long)(n0 + (lane & 15)) * ldw + k0 + 64, 0, 1);
        BFrag a = load_a_frag(sA, Kd, 0, k0, lane, 16);
        acc = __builtin_amdgcn_wmma_f32_16x16x32_bf16(false, a.v, false, bcur.v,
                                                      (short)0, acc, false, false);
        bcur = bnext;
    }
    int col = n0 + (lane & 15);
#pragma unroll
    for (int v2 = 0; v2 < 8; ++v2) {
        int row = row0 + v2 + ((lane >> 4) << 3);
        if (row < M) {
            float val = acc[v2] * alpha;
            if (rowscale) val *= rowscale[row];
            Cmat[(long)row * ldc + col] = val;
        }
    }
}

// ---------------------------------------------------------------------------
// k_medge: per batch: transformed = gelu(XW + tpb[r]); M_edge = C@transformed/counts
// grid = B, block = 256
// ---------------------------------------------------------------------------
__global__ void k_medge(const float* __restrict__ XW, const float* __restrict__ tpb,
                        const float* __restrict__ Mctx, const float* __restrict__ counts,
                        unsigned short* __restrict__ Medge) {
    __shared__ __align__(16) float sXW[KK * DD];
    __shared__ float sW[KK][KK];
    int b = blockIdx.x, tid = threadIdx.x;
    for (int i = tid; i < (KK * DD) >> 2; i += 256)     // 16B chunks of fp32
        async_copy_b128(&sXW[i << 2], XW + (long)b * KK * DD + (i << 2));
    if (tid < KK * KK) {
        int s = tid / KK, t = tid % KK;
        float m = (t != s) ? Mctx[b * KK + t] : 0.f;
        sW[s][t] = m / counts[b * KK + s];
    }
    wait_async0();
    __syncthreads();
#pragma unroll
    for (int dd = 0; dd < 2; ++dd) {
        int d = tid + dd * 256;
#pragma unroll
        for (int r = 0; r < RR; ++r) {
            float tr[KK];
            float tb = tpb[r * DD + d];
#pragma unroll
            for (int t = 0; t < KK; ++t) tr[t] = gelu_exact(sXW[t * DD + d] + tb);
#pragma unroll
            for (int s = 0; s < KK; ++s) {
                float a = 0.f;
#pragma unroll
                for (int t = 0; t < KK; ++t) a += sW[s][t] * tr[t];
                Medge[(((long)(b * KK + s)) * RR + r) * DD + d] = f2bf(a);
            }
        }
    }
}

// ---------------------------------------------------------------------------
// k_router: FiLM router -> route_logits
// grid = (K, B), block = 64
// ---------------------------------------------------------------------------
__global__ void k_router(const float* __restrict__ pooled, const float* __restrict__ Qbar,
                         const float* __restrict__ ctx_w, const float* __restrict__ ctx_b,
                         const float* __restrict__ gamma_w, const float* __restrict__ gamma_b,
                         const float* __restrict__ beta_w, const float* __restrict__ beta_b,
                         const float* __restrict__ rout_w, const float* __restrict__ rout_b,
                         float* __restrict__ rlog) {
    __shared__ float sMod[CTX];
    int k = blockIdx.x, b = blockIdx.y, c = threadIdx.x;
    const float* pp = pooled + ((long)(b * KK + k)) * DD;
    const float* cw = ctx_w + ((long)(k * CTX + c)) * DD;
    float a = ctx_b[k * CTX + c];
#pragma unroll 8
    for (int d = 0; d < DD; ++d) a += pp[d] * cw[d];
    float q = Qbar[b * KK + k];
    float g = q * gamma_w[k * CTX + c] + gamma_b[k * CTX + c];
    float be = q * beta_w[k * CTX + c] + beta_b[k * CTX + c];
    float mod = g * a + be;
    sMod[c] = (mod > 0.f) ? mod : 0.f;
    __syncthreads();
    if (c < RR) {
        float s = rout_b[k * RR + c];
        const float* rw = rout_w + (long)(k * RR + c) * CTX;
#pragma unroll 8
        for (int j = 0; j < CTX; ++j) s += sMod[j] * rw[j];
        rlog[((long)(b * KK + k)) * RR + c] = s;
    }
}

// ---------------------------------------------------------------------------
// k_attn: fused edge-token attention. One block = 2 batches x (R=8 rows) = 16 WMMA rows.
// Pass 1: Ke = M_edge @ Wke^T -> logits (LDS ds_add_f32 reduction) -> softmax
// Pass 2: Ve = M_edge @ Wve^T -> out = beta-weighted sum -> bf16
// grid = (K, B/2), block = 256 (8 waves, 4 n-tiles each)
// ---------------------------------------------------------------------------
__global__ void k_attn(const unsigned short* __restrict__ Medge,
                       const float* __restrict__ Qh,      // pre-scaled by 1/sqrt(dh)
                       const float* __restrict__ rlog,
                       const unsigned short* __restrict__ Wke,
                       const unsigned short* __restrict__ Wve,
                       unsigned short* __restrict__ attn_out) {
    __shared__ __align__(16) unsigned short sA[16 * DD];  // rows m = bb*8 + r
    __shared__ __align__(16) float sQ[2 * DD];
    __shared__ float sLog[2][HH][RR];
    __shared__ float sBeta[2][HH][RR];
    __shared__ float sOut[2 * DD];

    int k = blockIdx.x;
    int b0 = blockIdx.y * 2;
    int tid = threadIdx.x;
    int lane = tid & 31;
    int wave = tid >> 5;

    // ---- async stage: M_edge tile (16KB) + Qh rows (4KB) ----
    for (int i = tid; i < (16 * DD) >> 3; i += 256) {
        int idx = i << 3;
        int m = idx >> 9, d = idx & (DD - 1);
        int bb = m >> 3, r = m & 7;
        async_copy_b128(&sA[idx],
                        Medge + (((long)((b0 + bb) * KK + k)) * RR + r) * DD + d);
    }
    {
        int idx = tid << 2;                 // 256 threads x 4 floats = 2*DD
        int bb = idx >> 9, d = idx & (DD - 1);
        async_copy_b128(&sQ[idx], Qh + ((long)((b0 + bb) * KK + k)) * DD + d);
    }
    for (int i = tid; i < 2 * DD; i += 256) sOut[i] = 0.f;
    if (tid < 2 * HH * RR) ((float*)sLog)[tid] = 0.f;
    wait_async0();
    __syncthreads();

    // ---- Pass 1: Ke, reduce into logits ----
#pragma unroll
    for (int it = 0; it < 4; ++it) {
        int n0 = (wave * 4 + it) * 16;
        v8f acc = {};
        BFrag bcur = load_b_frag(Wke, DD, n0, 0, lane);
        for (int k0 = 0; k0 < DD; k0 += 32) {
            int kn = (k0 + 32 < DD) ? (k0 + 32) : 0;
            BFrag bnext = load_b_frag(Wke, DD, n0, kn, lane);
            BFrag a = load_a_frag(sA, DD, 0, k0, lane, 16);
            acc = __builtin_amdgcn_wmma_f32_16x16x32_bf16(false, a.v, false, bcur.v,
                                                          (short)0, acc, false, false);
            bcur = bnext;
        }
        int bbl = lane >> 4;
        int n = n0 + (lane & 15);
        int h = n >> 8;   // DH = 256
        float q = sQ[bbl * DD + n];
#pragma unroll
        for (int v2 = 0; v2 < 8; ++v2)
            atomicAdd(&sLog[bbl][h][v2], acc[v2] * q);
    }
    __syncthreads();

    // ---- softmax over R per (bb, h), with route_logits added ----
    if (tid < 2 * HH) {
        int bb = tid >> 1, h = tid & 1;
        int b = b0 + bb;
        float l[RR], mx = -3.0e38f;
#pragma unroll
        for (int r = 0; r < RR; ++r) {
            l[r] = sLog[bb][h][r] + rlog[((long)(b * KK + k)) * RR + r];
            if (l[r] > mx) mx = l[r];
        }
        float s = 0.f;
#pragma unroll
        for (int r = 0; r < RR; ++r) { l[r] = __expf(l[r] - mx); s += l[r]; }
        float inv = (s > 0.f) ? (1.f / s) : 0.f;   // nan_to_num guard
#pragma unroll
        for (int r = 0; r < RR; ++r) sBeta[bb][h][r] = l[r] * inv;
    }
    __syncthreads();

    // ---- Pass 2: Ve, beta-weighted accumulation ----
#pragma unroll
    for (int it = 0; it < 4; ++it) {
        int n0 = (wave * 4 + it) * 16;
        v8f acc = {};
        BFrag bcur = load_b_frag(Wve, DD, n0, 0, lane);
        for (int k0 = 0; k0 < DD; k0 += 32) {
            int kn = (k0 + 32 < DD) ? (k0 + 32) : 0;
            BFrag bnext = load_b_frag(Wve, DD, n0, kn, lane);
            BFrag a = load_a_frag(sA, DD, 0, k0, lane, 16);
            acc = __builtin_amdgcn_wmma_f32_16x16x32_bf16(false, a.v, false, bcur.v,
                                                          (short)0, acc, false, false);
            bcur = bnext;
        }
        int bbl = lane >> 4;
        int n = n0 + (lane & 15);
        int h = n >> 8;
        float o = 0.f;
#pragma unroll
        for (int v2 = 0; v2 < 8; ++v2) o += sBeta[bbl][h][v2] * acc[v2];
        sOut[bbl * DD + n] = o;   // unique (bb, n) writer per block
    }
    __syncthreads();

    for (int i = tid; i < 2 * DD; i += 256) {
        int bb = i >> 9, d = i & (DD - 1);
        attn_out[((long)((b0 + bb) * KK + k)) * DD + d] = f2bf(sOut[i]);
    }
}

// ---------------------------------------------------------------------------
// Host launcher
// ---------------------------------------------------------------------------
extern "C" void kernel_launch(void* const* d_in, const int* in_sizes, int n_in,
                              void* d_out, int out_size, void* d_ws, size_t ws_size,
                              hipStream_t stream) {
    const float* X        = (const float*)d_in[0];
    const float* Mctx     = (const float*)d_in[1];
    const float* Qbar     = (const float*)d_in[2];
    const float* qseed    = (const float*)d_in[3];
    const float* etok     = (const float*)d_in[4];
    const float* W1       = (const float*)d_in[5];
    const float* b1       = (const float*)d_in[6];
    const float* Wq       = (const float*)d_in[7];
    const float* Wke      = (const float*)d_in[8];
    const float* Wve      = (const float*)d_in[9];
    const float* Wo       = (const float*)d_in[10];
    const float* ctx_w    = (const float*)d_in[11];
    const float* ctx_b    = (const float*)d_in[12];
    const float* gamma_w  = (const float*)d_in[13];
    const float* gamma_b  = (const float*)d_in[14];
    const float* beta_w   = (const float*)d_in[15];
    const float* beta_b   = (const float*)d_in[16];
    const float* rout_w   = (const float*)d_in[17];
    const float* rout_b   = (const float*)d_in[18];
    float* Z = (float*)d_out;

    // workspace carve-out
    char* p = (char*)d_ws;
    auto take = [&](size_t bytes) -> char* {
        char* q = p;
        p += (bytes + 255) & ~size_t(255);
        return q;
    };
    const long BK = (long)BB * KK;                 // 12288
    float*          counts   = (float*)take(BK * 4);
    float*          has_src  = (float*)take(BK * 4);
    float*          pooled   = (float*)take(BK * DD * 4);
    unsigned short* Xbf      = (unsigned short*)take(BK * DD * 2);
    unsigned short* qsbf     = (unsigned short*)take(BK * DD * 2);
    unsigned short* W1x_bf   = (unsigned short*)take((long)DD * DD * 2);
    unsigned short* Wq_bf    = (unsigned short*)take((long)DD * DD * 2);
    unsigned short* Wke_bf   = (unsigned short*)take((long)DD * DD * 2);
    unsigned short* Wve_bf   = (unsigned short*)take((long)DD * DD * 2);
    unsigned short* Wo_bf    = (unsigned short*)take((long)DD * DD * 2);
    float*          tpb      = (float*)take((long)RR * DD * 4);
    float*          XW       = (float*)take(BK * DD * 4);
    unsigned short* Medge    = (unsigned short*)take(BK * RR * DD * 2);
    float*          QhS      = (float*)take(BK * DD * 4);
    float*          rlog     = (float*)take(BK * RR * 4);
    unsigned short* attn_bf  = (unsigned short*)take(BK * DD * 2);
    (void)ws_size; (void)in_sizes; (void)n_in; (void)out_size;

    // 1. prep: counts / has_src / pooled / X->bf16
    k_prep<<<BB, 256, 0, stream>>>(X, Mctx, counts, has_src, pooled, Xbf);

    // 2. conversions to bf16
    {
        long n = BK * DD;
        k_conv2d<<<(int)((n + 255) / 256), 256, 0, stream>>>(qseed, qsbf, (int)BK, DD, DD);
        long nw = (long)DD * DD;
        int gw = (int)((nw + 255) / 256);
        k_conv2d<<<gw, 256, 0, stream>>>(W1, W1x_bf, DD, DD, DD + EE); // W1[:, :D]
        k_conv2d<<<gw, 256, 0, stream>>>(Wq,  Wq_bf,  DD, DD, DD);
        k_conv2d<<<gw, 256, 0, stream>>>(Wke, Wke_bf, DD, DD, DD);
        k_conv2d<<<gw, 256, 0, stream>>>(Wve, Wve_bf, DD, DD, DD);
        k_conv2d<<<gw, 256, 0, stream>>>(Wo,  Wo_bf,  DD, DD, DD);
    }

    // 3. tpb = edge_tokens @ W1[:, D:].T + b1
    k_tpb<<<RR, 256, 0, stream>>>(etok, W1, b1, tpb);

    // 4. XW = X @ W1x^T   [12288,512]
    {
        dim3 grid((DD + 127) / 128, (int)((BK + 15) / 16));
        k_gemm_bf16<<<grid, 256, 0, stream>>>(Xbf, DD, W1x_bf, DD, XW, DD,
                                              (int)BK, DD, DD, 1.0f, nullptr);
    }

    // 5. M_edge (fused gelu + leave-one-out reduction), bf16
    k_medge<<<BB, 256, 0, stream>>>(XW, tpb, Mctx, counts, Medge);

    // 6. FiLM router -> route_logits
    k_router<<<dim3(KK, BB), CTX, 0, stream>>>(pooled, Qbar, ctx_w, ctx_b,
                                               gamma_w, gamma_b, beta_w, beta_b,
                                               rout_w, rout_b, rlog);

    // 7. Qh = query_seed @ Wq^T, pre-scaled by 1/sqrt(d_h) = 1/16
    {
        dim3 grid((DD + 127) / 128, (int)((BK + 15) / 16));
        k_gemm_bf16<<<grid, 256, 0, stream>>>(qsbf, DD, Wq_bf, DD, QhS, DD,
                                              (int)BK, DD, DD, 0.0625f, nullptr);
    }

    // 8. fused Ke/Ve attention (2 batches per block -> full 16 WMMA rows)
    k_attn<<<dim3(KK, BB / 2), 256, 0, stream>>>(Medge, QhS, rlog, Wke_bf, Wve_bf, attn_bf);

    // 9. Z = (attn @ Wo^T) * has_src  -> d_out (fp32)
    {
        dim3 grid((DD + 127) / 128, (int)((BK + 15) / 16));
        k_gemm_bf16<<<grid, 256, 0, stream>>>(attn_bf, DD, Wo_bf, DD, Z, DD,
                                              (int)BK, DD, DD, 1.0f, has_src);
    }
}
